// MultiheadAttention_21363167330835
// MI455X (gfx1250) — matrix operand order repaired and checked by reading, not compiled, not run on hardware
//
#include <hip/hip_runtime.h>
#include <hip/hip_bf16.h>

// ---------------- problem constants (from the reference) ----------------
#define T_LEN 4096
#define BATCH 2
#define E_DIM 1024
#define HEADS 16
#define HD    64
#define BLK   512
#define HALF  256
#define NBKS  16                     // T_LEN / HALF
#define NBLK  (NBKS * BATCH * HEADS) // 512 attention blocks

typedef _Float16 half16 __attribute__((ext_vector_type(16)));
typedef _Float16 half8  __attribute__((ext_vector_type(8)));
typedef _Float16 half4  __attribute__((ext_vector_type(4)));
typedef _Float16 half2  __attribute__((ext_vector_type(2)));
typedef float    float8 __attribute__((ext_vector_type(8)));
typedef float    float4v __attribute__((ext_vector_type(4)));
typedef float    float2v __attribute__((ext_vector_type(2)));

// gfx1250 async global->LDS copy path (ASYNCcnt-tracked, bypasses VGPRs).
// Builtin signature (from hipcc diagnostics): param0 = v4i addrspace(1)*,
// param1 = v4i addrspace(3)*, then two integer-constant args (offset, cpol).
#if defined(__gfx1250__) && \
    __has_builtin(__builtin_amdgcn_global_load_async_to_lds_b128) && \
    __has_builtin(__builtin_amdgcn_s_wait_asynccnt)
#define ASYNC_LDS 1
typedef int int4g __attribute__((vector_size(16)));
#define AS1(p) ((__attribute__((address_space(1))) int4g*)(p))
#define AS3(p) ((__attribute__((address_space(3))) int4g*)(p))
#endif

// ---------------- WMMA fragment loaders (wave32, 16x16x32 f16) ----------
// A (16x32, MxK): lane l holds row m=l&15; elem i -> k=(l>>4)*16+i
// (contiguous order, matching frag_b)
__device__ inline half16 frag_a(const _Float16* s, int ld, int m0, int k0, int lane) {
  const _Float16* p = s + (m0 + (lane & 15)) * ld + k0 + ((lane >> 4) << 4);
  half16 a;
#pragma unroll
  for (int i = 0; i < 16; ++i) a[i] = p[i];
  return a;
}
// B (32x16, KxN) staged in LDS as [n][k]: lane l holds col n=l&15,
// elem i -> k=(l>>4)*16+i  => one contiguous 32B read per lane
__device__ inline half16 frag_b(const _Float16* s, int ld, int n0, int k0, int lane) {
  const _Float16* p = s + (n0 + (lane & 15)) * ld + k0 + ((lane >> 4) << 4);
  half16 b;
#pragma unroll
  for (int i = 0; i < 16; ++i) b[i] = p[i];
  return b;
}

// 8 contiguous elements -> half8 (converting f32 -> f16 when needed)
__device__ inline half8 load8_cvt(const float* p) {
  const float4v a = *(const float4v*)p;
  const float4v b = *(const float4v*)(p + 4);
  const half4 ha = __builtin_convertvector(a, half4);
  const half4 hb = __builtin_convertvector(b, half4);
  half8 r;
#pragma unroll
  for (int i = 0; i < 4; ++i) { r[i] = ha[i]; r[i + 4] = hb[i]; }
  return r;
}
__device__ inline half8 load8_cvt(const _Float16* p) { return *(const half8*)p; }

// ---------------- generic Y = X @ W^T + bias (scaled) GEMM --------------
// X: MxK (f32 or f16), W: NxK f32 (rows feed B frags directly), Y: MxN f32.
// Block: 64x64 tile, 8 waves x two 16x16 tiles, K staged 32 at a time.
template <typename TA>
__global__ void __launch_bounds__(256)
gemm_bias_wmma(const TA* __restrict__ X, const float* __restrict__ W,
               const float* __restrict__ bias, float* __restrict__ Y,
               int M, int N, int K, float scale) {
  __shared__ _Float16 As[64 * 40];
  __shared__ _Float16 Bs[64 * 40];
  const int tid = threadIdx.x, lane = tid & 31, wave = tid >> 5;
  const int n0 = blockIdx.x * 64, m0 = blockIdx.y * 64;
  const int tr = (wave >> 1) * 16;   // wave tile row in block tile
  const int tc = (wave & 1) * 32;    // wave covers cols [tc, tc+32)
  const int lrow = tid >> 2;         // 0..63
  const int lcol = (tid & 3) * 8;    // 0,8,16,24
  float8 acc0 = {}, acc1 = {};
  for (int k0 = 0; k0 < K; k0 += 32) {
    // b128 global loads, b128 LDS stores (f32->f16 convert in flight)
    *(half8*)&As[lrow * 40 + lcol] = load8_cvt(X + (long)(m0 + lrow) * K + k0 + lcol);
    *(half8*)&Bs[lrow * 40 + lcol] = load8_cvt(W + (long)(n0 + lrow) * K + k0 + lcol);
    __syncthreads();
    half16 a  = frag_a(As, 40, tr, 0, lane);
    half16 b0 = frag_b(Bs, 40, tc, 0, lane);
    half16 b1 = frag_b(Bs, 40, tc + 16, 0, lane);
    acc0 = __builtin_amdgcn_wmma_f32_16x16x32_f16(false, a, false, b0, (short)0, acc0, false, false);
    acc1 = __builtin_amdgcn_wmma_f32_16x16x32_f16(false, a, false, b1, (short)0, acc1, false, false);
    __syncthreads();
  }
  const int mrow = m0 + tr + ((lane >> 4) << 3);
  const int ncol = n0 + tc + (lane & 15);
#pragma unroll
  for (int r = 0; r < 8; ++r) {
    const long m = mrow + r;
    Y[m * N + ncol]      = (acc0[r] + bias[ncol])      * scale;
    Y[m * N + ncol + 16] = (acc1[r] + bias[ncol + 16]) * scale;
  }
}

// ---------------- RoPE / windowing (pairwise, f32 -> f16) ---------------
// GPT-J rotate-every-two: out[2p]   = x[2p]*c - x[2p+1]*s
//                         out[2p+1] = x[2p+1]*c + x[2p]*s
__global__ void rope_q_kernel(const float* __restrict__ qlin,
                              const float* __restrict__ sin_t,
                              const float* __restrict__ cos_t,
                              const float* __restrict__ scale,
                              _Float16* __restrict__ qf) {
  const int idx = blockIdx.x * blockDim.x + threadIdx.x; // NBLK*HALF*32
  const int p = idx & 31, r = (idx >> 5) & 255, n = idx >> 13;
  const int h = n & 15, b = (n >> 4) & 1, nb = n >> 5;
  const long src = ((long)(nb * HALF + r) * BATCH + b) * E_DIM + h * HD + 2 * p;
  const float2v x = *(const float2v*)(qlin + src);
  const int pos = HALF + r;
  const float s  = scale[pos];
  const float c  = cos_t[pos * 32 + p] * s;
  const float sn = sin_t[pos * 32 + p] * s;
  half2 o;
  o[0] = (_Float16)(x[0] * c - x[1] * sn);
  o[1] = (_Float16)(x[1] * c + x[0] * sn);
  *(half2*)(qf + ((long)(n * HALF + r) * HD + 2 * p)) = o;
}

__global__ void rope_k_kernel(const float* __restrict__ klin,
                              const float* __restrict__ sin_t,
                              const float* __restrict__ cos_t,
                              const float* __restrict__ scale,
                              _Float16* __restrict__ kf) {
  const int idx = blockIdx.x * blockDim.x + threadIdx.x; // NBLK*BLK*32
  const int p = idx & 31, j = (idx >> 5) & 511, n = idx >> 14;
  const int h = n & 15, b = (n >> 4) & 1, nb = n >> 5;
  const int t = (nb - 1) * HALF + j;        // window: 256 history + 256 new
  half2 o = {};
  if (t >= 0) {
    const long src = ((long)t * BATCH + b) * E_DIM + h * HD + 2 * p;
    const float2v x = *(const float2v*)(klin + src);
    const float s  = 1.0f / scale[j];
    const float c  = cos_t[j * 32 + p] * s;
    const float sn = sin_t[j * 32 + p] * s;
    o[0] = (_Float16)(x[0] * c - x[1] * sn);
    o[1] = (_Float16)(x[1] * c + x[0] * sn);
  }
  *(half2*)(kf + ((long)(n * BLK + j) * HD + 2 * p)) = o;
}

__global__ void win_v_kernel(const float* __restrict__ vlin, _Float16* __restrict__ vf) {
  const int idx = blockIdx.x * blockDim.x + threadIdx.x; // NBLK*BLK*8
  const int g = idx & 7, j = (idx >> 3) & 511, n = idx >> 12;
  const int h = n & 15, b = (n >> 4) & 1, nb = n >> 5;
  const int t = (nb - 1) * HALF + j;
  half8 o = {};
  if (t >= 0) o = load8_cvt(vlin + ((long)t * BATCH + b) * E_DIM + h * HD + g * 8);
  *(half8*)(vf + ((long)(n * BLK + j) * HD + g * 8)) = o;
}

// ---------------- scores = q @ k^T + mask, written in probs layout ------
__global__ void __launch_bounds__(256)
scores_wmma(const _Float16* __restrict__ qf, const _Float16* __restrict__ kf,
            const float* __restrict__ mask, float* __restrict__ probs) {
  __shared__ _Float16 As[64 * 72];
  __shared__ _Float16 Bs[64 * 72];
  const int n = blockIdx.z, q0 = blockIdx.y * 64, j0 = blockIdx.x * 64;
  const int tid = threadIdx.x, lane = tid & 31, wave = tid >> 5;
  const int lrow = tid >> 2, lcol = (tid & 3) * 16;
  const _Float16* qg = qf + ((long)n * HALF + q0 + lrow) * HD + lcol;
  const _Float16* kg = kf + ((long)n * BLK  + j0 + lrow) * HD + lcol;
#ifdef ASYNC_LDS
  // Pure f16 byte copy -> use the async global->LDS path (ASYNCcnt).
  __builtin_amdgcn_global_load_async_to_lds_b128(AS1(qg),     AS3(&As[lrow * 72 + lcol]),     0, 0);
  __builtin_amdgcn_global_load_async_to_lds_b128(AS1(qg + 8), AS3(&As[lrow * 72 + lcol + 8]), 0, 0);
  __builtin_amdgcn_global_load_async_to_lds_b128(AS1(kg),     AS3(&Bs[lrow * 72 + lcol]),     0, 0);
  __builtin_amdgcn_global_load_async_to_lds_b128(AS1(kg + 8), AS3(&Bs[lrow * 72 + lcol + 8]), 0, 0);
  __builtin_amdgcn_s_wait_asynccnt(0);
#else
  *(half8*)&As[lrow * 72 + lcol]     = *(const half8*)qg;
  *(half8*)&As[lrow * 72 + lcol + 8] = *(const half8*)(qg + 8);
  *(half8*)&Bs[lrow * 72 + lcol]     = *(const half8*)kg;
  *(half8*)&Bs[lrow * 72 + lcol + 8] = *(const half8*)(kg + 8);
#endif
  __syncthreads();
  const int tr = (wave >> 1) * 16, tc = (wave & 1) * 32;
  float8 acc0 = {}, acc1 = {};
#pragma unroll
  for (int k0 = 0; k0 < HD; k0 += 32) {
    half16 a  = frag_a(As, 72, tr, k0, lane);
    half16 b0 = frag_b(Bs, 72, tc, k0, lane);
    half16 b1 = frag_b(Bs, 72, tc + 16, k0, lane);
    acc0 = __builtin_amdgcn_wmma_f32_16x16x32_f16(false, a, false, b0, (short)0, acc0, false, false);
    acc1 = __builtin_amdgcn_wmma_f32_16x16x32_f16(false, a, false, b1, (short)0, acc1, false, false);
  }
  // probs layout: [HEADS][NBKS*BATCH][HALF][BLK]
  const int h = n & 15, nbb = n >> 4;
  float* base = probs + ((long)h * (NBKS * BATCH) + nbb) * HALF * BLK;
  const int qm = q0 + tr + ((lane >> 4) << 3);
  const int jn = j0 + tc + (lane & 15);
#pragma unroll
  for (int r = 0; r < 8; ++r) {
    const int q = qm + r;
    base[(long)q * BLK + jn]      = acc0[r] + mask[q * BLK + jn];
    base[(long)q * BLK + jn + 16] = acc1[r] + mask[q * BLK + jn + 16];
  }
}

// ---------------- in-place row softmax over BLK=512 ---------------------
__global__ void __launch_bounds__(256)
softmax_rows(float* __restrict__ probs) {
  __shared__ float red[256];
  float* row = probs + (long)blockIdx.x * BLK;
  const int tid = threadIdx.x;
  const float a = row[tid], b = row[tid + 256];
  red[tid] = fmaxf(a, b);
  __syncthreads();
  for (int s = 128; s > 0; s >>= 1) {
    if (tid < s) red[tid] = fmaxf(red[tid], red[tid + s]);
    __syncthreads();
  }
  const float m = red[0];
  __syncthreads();
  const float ea = __expf(a - m), eb = __expf(b - m);
  red[tid] = ea + eb;
  __syncthreads();
  for (int s = 128; s > 0; s >>= 1) {
    if (tid < s) red[tid] += red[tid + s];
    __syncthreads();
  }
  const float inv = 1.0f / red[0];
  row[tid] = ea * inv;
  row[tid + 256] = eb * inv;
}

// ---------------- attn = probs @ v, scattered to (T,BATCH,E) f16 --------
__global__ void __launch_bounds__(256)
attnv_wmma(const float* __restrict__ probs, const _Float16* __restrict__ vf,
           _Float16* __restrict__ attn_ws) {
  __shared__ _Float16 As[64 * 72];
  __shared__ _Float16 Bs[64 * 72];   // stored transposed: Bs[d][j]
  const int n = blockIdx.y, q0 = blockIdx.x * 64;
  const int tid = threadIdx.x, lane = tid & 31, wave = tid >> 5;
  const int h = n & 15, b = (n >> 4) & 1, nb = n >> 5, nbb = n >> 4;
  const float* prow = probs + ((long)h * (NBKS * BATCH) + nbb) * HALF * BLK;
  const int lrow = tid >> 2, lcol = (tid & 3) * 16;
  const int tr = (wave >> 1) * 16, tc = (wave & 1) * 32;
  float8 acc0 = {}, acc1 = {};
  for (int j0 = 0; j0 < BLK; j0 += 64) {
    const float*    pg = prow + (long)(q0 + lrow) * BLK + j0 + lcol;
    const _Float16* vg = vf + ((long)n * BLK + j0 + lrow) * HD + lcol;
    *(half8*)&As[lrow * 72 + lcol]     = load8_cvt(pg);      // probs f32->f16
    *(half8*)&As[lrow * 72 + lcol + 8] = load8_cvt(pg + 8);
    const half8 v0 = *(const half8*)vg;
    const half8 v1 = *(const half8*)(vg + 8);
#pragma unroll
    for (int i = 0; i < 8; ++i) {                            // V -> [d][j]
      Bs[(lcol + i) * 72 + lrow]     = v0[i];
      Bs[(lcol + 8 + i) * 72 + lrow] = v1[i];
    }
    __syncthreads();
#pragma unroll
    for (int k0 = 0; k0 < 64; k0 += 32) {
      half16 a  = frag_a(As, 72, tr, k0, lane);
      half16 b0 = frag_b(Bs, 72, tc, k0, lane);
      half16 b1 = frag_b(Bs, 72, tc + 16, k0, lane);
      acc0 = __builtin_amdgcn_wmma_f32_16x16x32_f16(false, a, false, b0, (short)0, acc0, false, false);
      acc1 = __builtin_amdgcn_wmma_f32_16x16x32_f16(false, a, false, b1, (short)0, acc1, false, false);
    }
    __syncthreads();
  }
  const int qm = q0 + tr + ((lane >> 4) << 3);
  const int dn = tc + (lane & 15);
#pragma unroll
  for (int r = 0; r < 8; ++r) {
    const int q = qm + r;
    const long dst = ((long)(nb * HALF + q) * BATCH + b) * E_DIM + h * HD;
    attn_ws[dst + dn]      = (_Float16)acc0[r];
    attn_ws[dst + dn + 16] = (_Float16)acc1[r];
  }
}

// ---------------- host-side launch --------------------------------------
extern "C" void kernel_launch(void* const* d_in, const int* in_sizes, int n_in,
                              void* d_out, int out_size, void* d_ws, size_t ws_size,
                              hipStream_t stream) {
  (void)in_sizes; (void)n_in; (void)out_size; (void)ws_size;
  const float* query = (const float*)d_in[0];
  const float* key_t = (const float*)d_in[1];
  const float* value = (const float*)d_in[2];
  const float* Wq = (const float*)d_in[3];
  const float* bq = (const float*)d_in[4];
  const float* Wk = (const float*)d_in[5];
  const float* bk = (const float*)d_in[6];
  const float* Wv = (const float*)d_in[7];
  const float* bv = (const float*)d_in[8];
  const float* Wo = (const float*)d_in[9];
  const float* bo = (const float*)d_in[10];
  const float* sin_t = (const float*)d_in[11];
  const float* cos_t = (const float*)d_in[12];
  const float* scale = (const float*)d_in[13];
  const float* attn_mask = (const float*)d_in[14];

  char* ws = (char*)d_ws;
  float*    q_lin = (float*)(ws);                             // 32 MB
  float*    k_lin = (float*)(ws + ((size_t)32  << 20));       // 32 MB
  float*    v_lin = (float*)(ws + ((size_t)64  << 20));       // 32 MB
  _Float16* qf    = (_Float16*)(ws + ((size_t)96  << 20));    // 16 MB
  _Float16* kf    = (_Float16*)(ws + ((size_t)112 << 20));    // 32 MB
  _Float16* vf    = (_Float16*)(ws + ((size_t)144 << 20));    // 32 MB
  _Float16* attn_ws = (_Float16*)(ws);                        // reuses dead q_lin region

  float* out   = (float*)d_out;
  float* probs = out + (size_t)T_LEN * BATCH * E_DIM;

  const int M = T_LEN * BATCH;         // 8192
  const dim3 gg(E_DIM / 64, M / 64);   // (16, 128)

  // 1. projections (q scaled by HD^-0.5 = 0.125)
  gemm_bias_wmma<float><<<gg, 256, 0, stream>>>(query, Wq, bq, q_lin, M, E_DIM, E_DIM, 0.125f);
  gemm_bias_wmma<float><<<gg, 256, 0, stream>>>(key_t, Wk, bk, k_lin, M, E_DIM, E_DIM, 1.0f);
  gemm_bias_wmma<float><<<gg, 256, 0, stream>>>(value, Wv, bv, v_lin, M, E_DIM, E_DIM, 1.0f);

  // 2. window + RoPE into f16 attention operands
  rope_q_kernel<<<(NBLK * HALF * 32) / 256, 256, 0, stream>>>(q_lin, sin_t, cos_t, scale, qf);
  rope_k_kernel<<<(NBLK * BLK * 32) / 256, 256, 0, stream>>>(k_lin, sin_t, cos_t, scale, kf);
  win_v_kernel <<<(NBLK * BLK * 8) / 256, 256, 0, stream>>>(v_lin, vf);

  // 3. scores + mask directly into attn_weights layout in d_out
  scores_wmma<<<dim3(BLK / 64, HALF / 64, NBLK), 256, 0, stream>>>(qf, kf, attn_mask, probs);

  // 4. softmax in place (these normalized probs ARE the 2nd output)
  softmax_rows<<<NBLK * HALF, 256, 0, stream>>>(probs);

  // 5. attn = probs @ v  -> (T, BATCH, E) f16
  attnv_wmma<<<dim3(HALF / 64, NBLK), 256, 0, stream>>>(probs, vf, attn_ws);

  // 6. output projection
  gemm_bias_wmma<_Float16><<<gg, 256, 0, stream>>>(attn_ws, Wo, bo, out, M, E_DIM, E_DIM, 1.0f);
}
